// MultiSingleStreamBlockLoraProcessor_28561532519137
// MI455X (gfx1250) — compile-verified, matrix operand order checked by reading, not run on hardware
//
#include <hip/hip_runtime.h>

#define S_TOT   2176
#define DMODEL  3072
#define N3      9216      // 3*DMODEL
#define BLOCK_Q 1152      // S - COND
#define CONDN   1024
#define NHEADS  24
#define HD      128
#define RANK    16
#define KSTEPS  96        // DMODEL / 32

typedef __attribute__((ext_vector_type(16))) __bf16 v16bf;
typedef __attribute__((ext_vector_type(8)))  __bf16 bf8v;
typedef __attribute__((ext_vector_type(4)))  __bf16 bf4v;
typedef __attribute__((ext_vector_type(8)))  float  v8f;
typedef __attribute__((ext_vector_type(4)))  float  f4v;

static __device__ __forceinline__ v16bf make16(bf8v lo, bf8v hi) {
  v16bf r;
#pragma unroll
  for (int i = 0; i < 8; i++) { r[i] = lo[i]; r[i + 8] = hi[i]; }
  return r;
}

// 16-byte global->LDS async copy (GLOBAL_LOAD_ASYNC_TO_LDS_B128, ASYNCcnt).
// LDS byte address = low 32 bits of the generic pointer (aperture is in the
// high dword per the CDNA5 address-map), global address = full 64-bit VGPR pair.
static __device__ __forceinline__ void cp16_g2l(const __bf16* g, __bf16* l) {
  unsigned int       lds_off = (unsigned int)(uintptr_t)l;
  unsigned long long gaddr   = (unsigned long long)(uintptr_t)g;
  asm volatile("global_load_async_to_lds_b128 %0, %1, off"
               :: "v"(lds_off), "v"(gaddr)
               : "memory");
}

static __device__ __forceinline__ void wait_async_all() {
  asm volatile("s_wait_asynccnt 0" ::: "memory");
}

// ---------------------------------------------------------------------------
// 0) fp32 -> bf16 bulk convert (weights), 4 elements / thread
// ---------------------------------------------------------------------------
__global__ __launch_bounds__(256) void cvt_bf16_kernel(
    const float* __restrict__ src, __bf16* __restrict__ dst, int n)
{
  const int i = (blockIdx.x * 256 + threadIdx.x) * 4;
  if (i < n) {
    f4v v = *(const f4v*)(src + i);
    bf4v o;
#pragma unroll
    for (int j = 0; j < 4; j++) o[j] = (__bf16)v[j];
    *(bf4v*)(dst + i) = o;
  }
}

// ---------------------------------------------------------------------------
// 1) LoRA down-projection: T[m][s][j] = sum_d hidden[1152+s][d] * down_m[j][d]
// ---------------------------------------------------------------------------
__global__ __launch_bounds__(256) void lora_down_kernel(
    const float* __restrict__ hidden,
    const float* __restrict__ qd, const float* __restrict__ kd,
    const float* __restrict__ vd, float* __restrict__ T)
{
  const int s = blockIdx.x;
  const int m = blockIdx.y;
  const float* down = (m == 0) ? qd : (m == 1) ? kd : vd;
  const float* hrow = hidden + (size_t)(BLOCK_Q + s) * DMODEL;
  const int j      = threadIdx.x >> 4;
  const int lane16 = threadIdx.x & 15;
  const float* drow = down + (size_t)j * DMODEL;
  float acc = 0.f;
  for (int d = lane16; d < DMODEL; d += 16) acc += hrow[d] * drow[d];
  acc += __shfl_xor(acc, 1, 32);
  acc += __shfl_xor(acc, 2, 32);
  acc += __shfl_xor(acc, 4, 32);
  acc += __shfl_xor(acc, 8, 32);
  if (lane16 == 0) T[((size_t)m * CONDN + s) * RANK + j] = acc;
}

// ---------------------------------------------------------------------------
// 2) Fused QKV GEMM, block tile 64x128, K-step 32, double-buffered LDS.
//    A panel (64x32) staged sync w/ fp32->bf16 convert; B panel (32x128,
//    stored [n][k]) staged with GLOBAL_LOAD_ASYNC_TO_LDS_B128 from the
//    pre-converted bf16 weights. One workgroup barrier per K-step.
//    8 waves: wave w -> M-subtile (w&3), N-half (w>>2), 4 accumulators.
// ---------------------------------------------------------------------------
__global__ __launch_bounds__(256) void qkv_gemm_wmma(
    const float* __restrict__ hidden,
    const __bf16* __restrict__ Wbf,            // [9216][3072] bf16 (q,k,v stacked)
    const float* __restrict__ bq, const float* __restrict__ bk,
    const float* __restrict__ bv,
    const float* __restrict__ T,
    const float* __restrict__ q_up, const float* __restrict__ k_up,
    const float* __restrict__ v_up,
    __bf16* __restrict__ qkv)
{
  __shared__ __attribute__((aligned(16))) __bf16 lA[2][64 * 32];   //  8 KB
  __shared__ __attribute__((aligned(16))) __bf16 lB[2][128 * 32];  // 16 KB

  const int tid   = threadIdx.x;
  const int wv    = tid >> 5;
  const int lane  = tid & 31;
  const int col16 = lane & 15;
  const int half  = lane >> 4;
  const int ko    = half ? 8 : 0;
  const int msub  = wv & 3;       // 0..3  -> M subtile
  const int nhalf = wv >> 2;      // 0..1  -> N half (64 cols each)

  const int m0  = blockIdx.x * 64;
  const int gn0 = blockIdx.y * 128;
  const int mat = gn0 / DMODEL;   // uniform for whole block (3072 % 128 == 0)

  // ---- staging helpers -----------------------------------------------------
  auto stageA = [&](int ks, int buf) {  // 64x32 fp32 -> bf16, 8 elems/thread
    const int eo  = tid * 8;
    const int row = eo >> 5;
    const int kc  = eo & 31;
    const float* hp = hidden + (size_t)(m0 + row) * DMODEL + ks * 32 + kc;
    f4v x0 = *(const f4v*)hp;
    f4v x1 = *(const f4v*)(hp + 4);
    bf8v y;
#pragma unroll
    for (int i = 0; i < 4; i++) { y[i] = (__bf16)x0[i]; y[4 + i] = (__bf16)x1[i]; }
    *(bf8v*)&lA[buf][row * 32 + kc] = y;
  };
  auto stageB = [&](int ks, int buf) {  // 128 rows x 32 k bf16, 2x16B/thread
#pragma unroll
    for (int c = 0; c < 2; c++) {
      const int id  = tid * 2 + c;
      const int row = id >> 2;
      const int kc  = (id & 3) * 8;
      cp16_g2l(Wbf + (size_t)(gn0 + row) * DMODEL + ks * 32 + kc,
               &lB[buf][row * 32 + kc]);
    }
  };

  v8f acc[4];
#pragma unroll
  for (int j = 0; j < 4; j++) acc[j] = (v8f){};

  stageA(0, 0);
  stageB(0, 0);

  for (int i = 0; i < KSTEPS; i++) {
    const int cur = i & 1;
    const int nxt = cur ^ 1;
    wait_async_all();      // this wave's async fills of `cur` done
    __syncthreads();       // everyone's fills done; everyone done reading `nxt`
    if (i + 1 < KSTEPS) {  // overlap: stage next buffer while computing
      stageA(i + 1, nxt);
      stageB(i + 1, nxt);
    }

    const __bf16* ap = &lA[cur][(msub * 16 + col16) * 32 + ko];
    v16bf af = make16(*(const bf8v*)ap, *(const bf8v*)(ap + 16));
#pragma unroll
    for (int j = 0; j < 4; j++) {
      const __bf16* bp = &lB[cur][(nhalf * 64 + j * 16 + col16) * 32 + ko];
      v16bf bfrag = make16(*(const bf8v*)bp, *(const bf8v*)(bp + 16));
      acc[j] = __builtin_amdgcn_wmma_f32_16x16x32_bf16(false, af, false, bfrag,
                                                       (short)0, acc[j], false, false);
    }
  }

  // ---- epilogue: bias + LoRA-up for conditioned rows, store bf16 ----------
  const float* bias = (mat == 0) ? bq   : (mat == 1) ? bk   : bv;
  const float* up   = (mat == 0) ? q_up : (mat == 1) ? k_up : v_up;

#pragma unroll
  for (int j = 0; j < 4; j++) {
    const int nn = gn0 + nhalf * 64 + j * 16 + col16;
    const int ln = nn - mat * DMODEL;
    const float bia = bias[ln];
    float upv[RANK];
#pragma unroll
    for (int t = 0; t < RANK; t++) upv[t] = up[(size_t)ln * RANK + t];
#pragma unroll
    for (int r = 0; r < 8; r++) {
      const int srow = m0 + msub * 16 + r + 8 * half;
      float c = acc[j][r] + bia;
      if (srow >= BLOCK_Q) {
        const float* Trow = T + ((size_t)mat * CONDN + (srow - BLOCK_Q)) * RANK;
        float lacc = 0.f;
#pragma unroll
        for (int t = 0; t < RANK; t++) lacc += Trow[t] * upv[t];
        c += lacc;
      }
      qkv[(size_t)srow * N3 + nn] = (__bf16)c;
    }
  }
}

// ---------------------------------------------------------------------------
// 3) RMS-norm + RoPE per (position, head). Q pre-scaled by 1/sqrt(128).
// ---------------------------------------------------------------------------
__global__ __launch_bounds__(128) void qk_norm_rope(
    const __bf16* __restrict__ qkv, __bf16* __restrict__ outh,
    int nbase, int rows, const float* __restrict__ w,
    const float* __restrict__ cosc, const float* __restrict__ sinc,
    float oscale)
{
  const int s = blockIdx.x;
  const int h = blockIdx.y;
  const int d = threadIdx.x;
  __shared__ float sh[HD];
  __shared__ float red[4];

  float x = (float)qkv[(size_t)s * N3 + nbase + h * HD + d];
  float v = x * x;
  v += __shfl_xor(v, 1, 32);
  v += __shfl_xor(v, 2, 32);
  v += __shfl_xor(v, 4, 32);
  v += __shfl_xor(v, 8, 32);
  v += __shfl_xor(v, 16, 32);
  if ((threadIdx.x & 31) == 0) red[threadIdx.x >> 5] = v;
  __syncthreads();
  const float var = (red[0] + red[1] + red[2] + red[3]) * (1.0f / 128.0f);
  const float rs  = rsqrtf(var + 1e-6f);
  const float xn  = x * rs * w[d];
  sh[d] = xn;
  __syncthreads();
  const float xr = (d & 1) ? sh[d - 1] : -sh[d + 1];
  const float y  = (xn * cosc[(size_t)s * HD + d] + xr * sinc[(size_t)s * HD + d]) * oscale;
  outh[((size_t)h * rows + s) * HD + d] = (__bf16)y;
}

// ---------------------------------------------------------------------------
// 4) Pack V transposed: Vt[h][d][s]
// ---------------------------------------------------------------------------
__global__ __launch_bounds__(256) void v_pack(
    const __bf16* __restrict__ qkv, __bf16* __restrict__ Vt)
{
  const int s  = blockIdx.x * 256 + threadIdx.x;
  const int hd = blockIdx.y;
  if (s < S_TOT)
    Vt[(size_t)hd * S_TOT + s] = qkv[(size_t)s * N3 + 2 * DMODEL + hd];
}

// ---------------------------------------------------------------------------
// 5) Flash attention (no mask needed for kept rows).
// ---------------------------------------------------------------------------
__global__ __launch_bounds__(128) void attn_fa_wmma(
    const __bf16* __restrict__ Qh, const __bf16* __restrict__ Kh,
    const __bf16* __restrict__ Vt, float* __restrict__ out)
{
  __shared__ __attribute__((aligned(16))) __bf16 pshare[4][16 * 32];
  const int wv    = threadIdx.x >> 5;
  const int lane  = threadIdx.x & 31;
  const int col16 = lane & 15;
  const int half  = lane >> 4;
  const int ko    = half ? 8 : 0;
  const int head  = blockIdx.y;
  const int q0    = (blockIdx.x * 4 + wv) * 16;

  const __bf16* Qb = Qh + ((size_t)head * BLOCK_Q + q0) * HD;
  const __bf16* Kb = Kh + (size_t)head * S_TOT * HD;
  const __bf16* Vb = Vt + (size_t)head * HD * S_TOT;
  __bf16* P = pshare[wv];

  v16bf a[4];
#pragma unroll
  for (int kk = 0; kk < 4; kk++) {
    const __bf16* p = Qb + col16 * HD + kk * 32 + ko;
    a[kk] = make16(*(const bf8v*)p, *(const bf8v*)(p + 16));
  }

  v8f accd[8];
#pragma unroll
  for (int i = 0; i < 8; i++) accd[i] = (v8f){};
  float mrow[8], lrow[8];
#pragma unroll
  for (int r = 0; r < 8; r++) { mrow[r] = -3.0e38f; lrow[r] = 0.f; }

  for (int kt = 0; kt < S_TOT; kt += 32) {
    if (kt + 32 < S_TOT)
      __builtin_prefetch(Kb + (size_t)(kt + 32 + col16) * HD, 0, 1);

    v8f s0 = {}, s1 = {};
#pragma unroll
    for (int kk = 0; kk < 4; kk++) {
      const __bf16* kp0 = Kb + (size_t)(kt + col16) * HD + kk * 32 + ko;
      const __bf16* kp1 = kp0 + 16 * HD;
      v16bf bk0 = make16(*(const bf8v*)kp0, *(const bf8v*)(kp0 + 16));
      v16bf bk1 = make16(*(const bf8v*)kp1, *(const bf8v*)(kp1 + 16));
      s0 = __builtin_amdgcn_wmma_f32_16x16x32_bf16(false, a[kk], false, bk0,
                                                   (short)0, s0, false, false);
      s1 = __builtin_amdgcn_wmma_f32_16x16x32_bf16(false, a[kk], false, bk1,
                                                   (short)0, s1, false, false);
    }

#pragma unroll
    for (int r = 0; r < 8; r++) {
      float mx = fmaxf(s0[r], s1[r]);
      mx = fmaxf(mx, __shfl_xor(mx, 1, 32));
      mx = fmaxf(mx, __shfl_xor(mx, 2, 32));
      mx = fmaxf(mx, __shfl_xor(mx, 4, 32));
      mx = fmaxf(mx, __shfl_xor(mx, 8, 32));
      const float mn    = fmaxf(mrow[r], mx);
      const float alpha = __expf(mrow[r] - mn);
      mrow[r] = mn;
      const float p0 = __expf(s0[r] - mn);
      const float p1 = __expf(s1[r] - mn);
      const int prow = r + 8 * half;
      P[prow * 32 + col16]      = (__bf16)p0;
      P[prow * 32 + 16 + col16] = (__bf16)p1;
      float rs = p0 + p1;
      rs += __shfl_xor(rs, 1, 32);
      rs += __shfl_xor(rs, 2, 32);
      rs += __shfl_xor(rs, 4, 32);
      rs += __shfl_xor(rs, 8, 32);
      lrow[r] = lrow[r] * alpha + rs;
#pragma unroll
      for (int dt = 0; dt < 8; dt++) accd[dt][r] = accd[dt][r] * alpha;
    }

    asm volatile("s_wait_dscnt 0" ::: "memory");

    const __bf16* pp = P + col16 * 32 + ko;
    v16bf pa = make16(*(const bf8v*)pp, *(const bf8v*)(pp + 16));

#pragma unroll
    for (int dt = 0; dt < 8; dt++) {
      const __bf16* vp = Vb + (size_t)(dt * 16 + col16) * S_TOT + kt + ko;
      v16bf vb = make16(*(const bf8v*)vp, *(const bf8v*)(vp + 16));
      accd[dt] = __builtin_amdgcn_wmma_f32_16x16x32_bf16(false, pa, false, vb,
                                                         (short)0, accd[dt], false, false);
    }
  }

#pragma unroll
  for (int dt = 0; dt < 8; dt++) {
#pragma unroll
    for (int r = 0; r < 8; r++) {
      const int srow = q0 + r + 8 * half;
      out[(size_t)srow * DMODEL + head * HD + dt * 16 + col16] = accd[dt][r] / lrow[r];
    }
  }
}

// ---------------------------------------------------------------------------
extern "C" void kernel_launch(void* const* d_in, const int* in_sizes, int n_in,
                              void* d_out, int out_size, void* d_ws, size_t ws_size,
                              hipStream_t stream)
{
  (void)in_sizes; (void)n_in; (void)out_size; (void)ws_size;

  const float* hidden  = (const float*)d_in[0];
  const float* Wq      = (const float*)d_in[1];
  const float* bq      = (const float*)d_in[2];
  const float* Wk      = (const float*)d_in[3];
  const float* bk      = (const float*)d_in[4];
  const float* Wv      = (const float*)d_in[5];
  const float* bv      = (const float*)d_in[6];
  const float* q_down  = (const float*)d_in[7];
  const float* q_up    = (const float*)d_in[8];
  const float* k_down  = (const float*)d_in[9];
  const float* k_up    = (const float*)d_in[10];
  const float* v_down  = (const float*)d_in[11];
  const float* v_up    = (const float*)d_in[12];
  const float* normq_w = (const float*)d_in[13];
  const float* normk_w = (const float*)d_in[14];
  const float* ropecos = (const float*)d_in[15];
  const float* ropesin = (const float*)d_in[16];
  float* out = (float*)d_out;

  char* ws = (char*)d_ws;
  size_t o = 0;
  __bf16* Wbf = (__bf16*)(ws + o); o += (size_t)N3 * DMODEL * 2;          // 56.6 MB
  __bf16* qkv = (__bf16*)(ws + o); o += (size_t)S_TOT * N3 * 2;           // 40.1 MB
  float*  T   = (float*) (ws + o); o += (size_t)3 * CONDN * RANK * 4;     //  0.2 MB
  __bf16* Qhh = (__bf16*)(ws + o); o += (size_t)NHEADS * BLOCK_Q * HD * 2;//  7.1 MB
  __bf16* Khh = (__bf16*)(ws + o); o += (size_t)NHEADS * S_TOT * HD * 2;  // 13.4 MB
  __bf16* Vtt = (__bf16*)(ws + o);                                        // 13.4 MB

  const int wn = DMODEL * DMODEL;            // elements per weight matrix
  const int cvtBlocks = wn / 1024;           // 4 elems/thread, 256 threads
  cvt_bf16_kernel<<<cvtBlocks, 256, 0, stream>>>(Wq, Wbf + (size_t)0 * wn, wn);
  cvt_bf16_kernel<<<cvtBlocks, 256, 0, stream>>>(Wk, Wbf + (size_t)1 * wn, wn);
  cvt_bf16_kernel<<<cvtBlocks, 256, 0, stream>>>(Wv, Wbf + (size_t)2 * wn, wn);

  lora_down_kernel<<<dim3(CONDN, 3), 256, 0, stream>>>(hidden, q_down, k_down, v_down, T);

  qkv_gemm_wmma<<<dim3(S_TOT / 64, N3 / 128), 256, 0, stream>>>(
      hidden, Wbf, bq, bk, bv, T, q_up, k_up, v_up, qkv);

  const float qscale = 0.08838834764831845f;  // 1/sqrt(128)
  qk_norm_rope<<<dim3(BLOCK_Q, NHEADS), 128, 0, stream>>>(
      qkv, Qhh, 0, BLOCK_Q, normq_w, ropecos, ropesin, qscale);
  qk_norm_rope<<<dim3(S_TOT, NHEADS), 128, 0, stream>>>(
      qkv, Khh, DMODEL, S_TOT, normk_w, ropecos, ropesin, 1.0f);

  v_pack<<<dim3((S_TOT + 255) / 256, NHEADS * HD), 256, 0, stream>>>(qkv, Vtt);

  attn_fa_wmma<<<dim3(BLOCK_Q / 64, NHEADS), 128, 0, stream>>>(Qhh, Khh, Vtt, out);
}
